// Generator_90718299226269
// MI455X (gfx1250) — compile-verified, matrix operand order checked by reading, not run on hardware
//
#include <hip/hip_runtime.h>
#include <hip/hip_bf16.h>

// ---------------- problem constants ----------------
#define BATCH 512
#define NOISE 128
#define HDIM  1024
#define SEQ   150
#define VOC   4
#define KDIM  2048          // 2*H  (x | h concatenated)
#define N4H   4096          // 4*H  (i,f,g,o gates)

typedef __bf16 bf16;
typedef __attribute__((ext_vector_type(16))) __bf16 v16bf;
typedef __attribute__((ext_vector_type(8)))  float  v8f;
typedef __attribute__((ext_vector_type(4)))  unsigned int v4u;

union FragU { v16bf v; v4u u[2]; };

__device__ __forceinline__ bf16 f2bf(float f) {
    union { float f; unsigned u; } x; x.f = f;
    unsigned r = x.u + 0x7FFFu + ((x.u >> 16) & 1u);   // round-to-nearest-even
    union { unsigned short s; bf16 b; } y;
    y.s = (unsigned short)(r >> 16);
    return y.b;
}

// gfx1250 async copy: global (16B/lane) -> LDS, tracked by ASYNCcnt.
__device__ __forceinline__ void async_copy_b128(unsigned lds_off, const void* gaddr) {
    asm volatile("global_load_async_to_lds_b128 %0, %1, off"
                 :: "v"(lds_off), "v"(gaddr) : "memory");
}

// ---------------------------------------------------------------------------
// 1) Weight prep: Wcat[g][k] (bf16), k<H -> W_ih[g][k], else W_hh[g][k-H].
//    Also bias_cat[g] = b_ih[g] + b_hh[g].
// ---------------------------------------------------------------------------
__global__ void k_convert(const float* __restrict__ Wih, const float* __restrict__ Whh,
                          const float* __restrict__ bih, const float* __restrict__ bhh,
                          bf16* __restrict__ Wcat, float* __restrict__ bias) {
    size_t idx = (size_t)blockIdx.x * 256 + threadIdx.x;   // 0 .. 4096*2048-1
    size_t g = idx >> 11;
    size_t k = idx & 2047;
    float v = (k < HDIM) ? Wih[g * HDIM + k] : Whh[g * HDIM + (k - HDIM)];
    Wcat[idx] = f2bf(v);
    if (k == 0) bias[g] = bih[g] + bhh[g];
}

// ---------------------------------------------------------------------------
// 2) Initial state: h = noise @ W_h0^T + b_h0 ; c likewise.
//    xh[b][0..H-1] = 0 (x0), xh[b][H..2H-1] = bf16(h).  c stays fp32.
// ---------------------------------------------------------------------------
__global__ void k_init(const float* __restrict__ noise,
                       const float* __restrict__ Wh0, const float* __restrict__ bh0,
                       const float* __restrict__ Wc0, const float* __restrict__ bc0,
                       bf16* __restrict__ xh, float* __restrict__ cst) {
    int idx = blockIdx.x * 256 + threadIdx.x;      // 0 .. B*H-1
    int b = idx >> 10;
    int j = idx & (HDIM - 1);
    const float* nr = noise + (size_t)b * NOISE;
    const float* wh = Wh0 + (size_t)j * NOISE;
    const float* wc = Wc0 + (size_t)j * NOISE;
    float ah = bh0[j], ac = bc0[j];
    #pragma unroll 8
    for (int k = 0; k < NOISE; ++k) { float n = nr[k]; ah += n * wh[k]; ac += n * wc[k]; }
    cst[idx] = ac;
    xh[(size_t)b * KDIM + j]        = f2bf(0.0f);
    xh[(size_t)b * KDIM + HDIM + j] = f2bf(ah);
}

// ---------------------------------------------------------------------------
// 3) Gate GEMM: gates[B][4H] = xh[B][2H](bf16) @ Wcat[4H][2H]^T(bf16), f32 acc.
//    Block tile 128(M) x 128(N); 8 waves (2x4); wave tile 64x32; K chunk 32.
//    Double-buffered LDS filled by GLOBAL_LOAD_ASYNC_TO_LDS_B128 (ASYNCcnt),
//    consumed by v_wmma_f32_16x16x32_bf16.
// ---------------------------------------------------------------------------
#define LDA 40      // 32 bf16 + 8 pad (80B row stride: 16B aligned, bank-spread)
#define NKT (KDIM / 32)

__device__ __forceinline__ void issue_tile(const bf16* __restrict__ xh,
                                           const bf16* __restrict__ W,
                                           unsigned ldsA, unsigned ldsB,
                                           int mBase, int nBase, int k0, int t) {
    #pragma unroll
    for (int i = 0; i < 2; ++i) {
        int idx = t + i * 256;          // 0..511
        int row = idx >> 2;
        int c4  = (idx & 3) * 8;        // bf16 offset of the 16B chunk
        unsigned dA = ldsA + (unsigned)(row * LDA + c4) * 2u;
        unsigned dB = ldsB + (unsigned)(row * LDA + c4) * 2u;
        async_copy_b128(dA, xh + (size_t)(mBase + row) * KDIM + k0 + c4);
        async_copy_b128(dB, W  + (size_t)(nBase + row) * KDIM + k0 + c4);
    }
}

__global__ void __launch_bounds__(256)
k_gemm(const bf16* __restrict__ xh, const bf16* __restrict__ W,
       float* __restrict__ gates) {
    __shared__ bf16 As[2][128 * LDA];
    __shared__ bf16 Bs[2][128 * LDA];

    const int t     = threadIdx.x;
    const int nBase = blockIdx.x * 128;
    const int mBase = blockIdx.y * 128;
    const int lane  = t & 31;
    const int wave  = t >> 5;
    const int wm    = wave >> 2;        // 0..1 : M
    const int wn    = wave & 3;         // 0..3 : N
    const int lrow  = lane & 15;
    const int sel   = lane >> 4;        // half-wave select per ISA 16-bit layout

    const unsigned ldsA0 = (unsigned)(unsigned long long)(const void*)&As[0][0];
    const unsigned ldsA1 = (unsigned)(unsigned long long)(const void*)&As[1][0];
    const unsigned ldsB0 = (unsigned)(unsigned long long)(const void*)&Bs[0][0];
    const unsigned ldsB1 = (unsigned)(unsigned long long)(const void*)&Bs[1][0];

    v8f zero = {};
    v8f acc[4][2];
    #pragma unroll
    for (int i = 0; i < 4; ++i)
        #pragma unroll
        for (int j = 0; j < 2; ++j) acc[i][j] = zero;

    // prologue: tile 0 -> buffer 0 (4 async instructions per lane)
    issue_tile(xh, W, ldsA0, ldsB0, mBase, nBase, 0, t);

    for (int kt = 0; kt < NKT; ++kt) {
        // pipeline: start tile kt+1 into the other buffer
        if (kt + 1 < NKT) {
            issue_tile(xh, W, (kt & 1) ? ldsA0 : ldsA1, (kt & 1) ? ldsB0 : ldsB1,
                       mBase, nBase, (kt + 1) * 32, t);
            // 8 outstanding, drain to 4 => tile kt's copies have landed
            asm volatile("s_wait_asynccnt 0x4" ::: "memory");
        } else {
            asm volatile("s_wait_asynccnt 0x0" ::: "memory");
        }
        __syncthreads();   // all waves' copies for tile kt visible in LDS

        const bf16* Ab = &As[kt & 1][0];
        const bf16* Bb = &Bs[kt & 1][0];

        // per-lane fragments: lanes 0-15 take K {0..7,16..23}, lanes 16-31 {8..15,24..31}
        v16bf a[4], bb[2];
        #pragma unroll
        for (int i = 0; i < 4; ++i) {
            int r = wm * 64 + i * 16 + lrow;
            FragU u;
            u.u[0] = *(const v4u*)(Ab + r * LDA + 8 * sel);
            u.u[1] = *(const v4u*)(Ab + r * LDA + 16 + 8 * sel);
            a[i] = u.v;
        }
        #pragma unroll
        for (int j = 0; j < 2; ++j) {
            int r = wn * 32 + j * 16 + lrow;
            FragU u;
            u.u[0] = *(const v4u*)(Bb + r * LDA + 8 * sel);
            u.u[1] = *(const v4u*)(Bb + r * LDA + 16 + 8 * sel);
            bb[j] = u.v;
        }
        #pragma unroll
        for (int i = 0; i < 4; ++i)
            #pragma unroll
            for (int j = 0; j < 2; ++j)
                acc[i][j] = __builtin_amdgcn_wmma_f32_16x16x32_bf16(
                    false, a[i], false, bb[j], (short)0, acc[i][j], false, false);

        __syncthreads();   // everyone done reading buf[kt&1] before it is refilled
    }

    // epilogue: C layout — VGPR r, lanes 0-15: (M=r, N=lane); lanes 16-31: (M=r+8)
    #pragma unroll
    for (int i = 0; i < 4; ++i) {
        int gm0 = mBase + wm * 64 + i * 16 + 8 * sel;
        #pragma unroll
        for (int j = 0; j < 2; ++j) {
            int gn = nBase + wn * 32 + j * 16 + lrow;
            float* o = gates + (size_t)gm0 * N4H + gn;
            #pragma unroll
            for (int r = 0; r < 8; ++r) o[(size_t)r * N4H] = acc[i][j][r];
        }
    }
}

// ---------------------------------------------------------------------------
// 4) Fused cell: gates -> (c,h) update -> dropout -> logits(4) reduce ->
//    gumbel-softmax -> y out -> x_next = y @ W_fc -> bf16 state writeback.
//    One workgroup (256 threads) per batch row.
// ---------------------------------------------------------------------------
__global__ void __launch_bounds__(256)
k_cell(const float* __restrict__ gates, const float* __restrict__ bias,
       const float* __restrict__ Wfc, const float* __restrict__ bfc,
       const float* __restrict__ gumbel, const float* __restrict__ dropm,
       int s, float* __restrict__ cst, bf16* __restrict__ xh,
       float* __restrict__ out) {
    const int b = blockIdx.x;
    const int t = threadIdx.x;
    __shared__ float red[4][256];
    __shared__ float yv[4];

    const float* grow = gates + (size_t)b * N4H;
    const float* mrow = dropm + ((size_t)b * SEQ + s) * HDIM;
    float a0 = 0.f, a1 = 0.f, a2 = 0.f, a3 = 0.f;

    #pragma unroll
    for (int jj = 0; jj < 4; ++jj) {
        int j = t + jj * 256;
        float ig = grow[j]            + bias[j];
        float fg = grow[HDIM + j]     + bias[HDIM + j];
        float gg = grow[2 * HDIM + j] + bias[2 * HDIM + j];
        float og = grow[3 * HDIM + j] + bias[3 * HDIM + j];
        float si = 1.f / (1.f + __expf(-ig));
        float sf = 1.f / (1.f + __expf(-fg));
        float so = 1.f / (1.f + __expf(-og));
        float tg = tanhf(gg);
        size_t cidx = (size_t)b * HDIM + j;
        float c = sf * cst[cidx] + si * tg;
        cst[cidx] = c;
        float h = so * tanhf(c);
        xh[(size_t)b * KDIM + HDIM + j] = f2bf(h);      // h carry for next GEMM
        float m = h * mrow[j];
        a0 += m * Wfc[j];
        a1 += m * Wfc[HDIM + j];
        a2 += m * Wfc[2 * HDIM + j];
        a3 += m * Wfc[3 * HDIM + j];
    }
    red[0][t] = a0; red[1][t] = a1; red[2][t] = a2; red[3][t] = a3;
    __syncthreads();
    for (int off = 128; off >= 1; off >>= 1) {
        if (t < off) {
            #pragma unroll
            for (int v = 0; v < 4; ++v) red[v][t] += red[v][t + off];
        }
        __syncthreads();
    }
    if (t == 0) {
        const float* gum = gumbel + ((size_t)b * SEQ + s) * VOC;
        float l[4], mx = -1e30f;
        #pragma unroll
        for (int v = 0; v < 4; ++v) { l[v] = red[v][0] + bfc[v] + gum[v]; mx = fmaxf(mx, l[v]); }
        float sum = 0.f;
        #pragma unroll
        for (int v = 0; v < 4; ++v) { l[v] = __expf(l[v] - mx); sum += l[v]; }
        float inv = 1.f / sum;
        #pragma unroll
        for (int v = 0; v < 4; ++v) {
            float y = l[v] * inv;
            yv[v] = y;
            out[((size_t)b * SEQ + s) * VOC + v] = y;
        }
    }
    __syncthreads();
    float y0 = yv[0], y1 = yv[1], y2 = yv[2], y3 = yv[3];
    #pragma unroll
    for (int jj = 0; jj < 4; ++jj) {
        int j = t + jj * 256;
        float x = y0 * Wfc[j] + y1 * Wfc[HDIM + j] +
                  y2 * Wfc[2 * HDIM + j] + y3 * Wfc[3 * HDIM + j];
        xh[(size_t)b * KDIM + j] = f2bf(x);             // x carry for next GEMM
    }
}

// ---------------------------------------------------------------------------
extern "C" void kernel_launch(void* const* d_in, const int* in_sizes, int n_in,
                              void* d_out, int out_size, void* d_ws, size_t ws_size,
                              hipStream_t stream) {
    const float* noise  = (const float*)d_in[0];
    const float* W_h0   = (const float*)d_in[1];
    const float* b_h0   = (const float*)d_in[2];
    const float* W_c0   = (const float*)d_in[3];
    const float* b_c0   = (const float*)d_in[4];
    const float* W_ih   = (const float*)d_in[5];
    const float* W_hh   = (const float*)d_in[6];
    const float* b_ih   = (const float*)d_in[7];
    const float* b_hh   = (const float*)d_in[8];
    const float* W_fc   = (const float*)d_in[9];
    const float* b_fc   = (const float*)d_in[10];
    const float* gumbel = (const float*)d_in[11];
    const float* dropm  = (const float*)d_in[12];

    char* ws = (char*)d_ws;
    bf16*  xh    = (bf16*)(ws);                             //  2 MB  [B][2H] bf16
    bf16*  Wcat  = (bf16*)(ws + (size_t)2  * (1 << 20));    // 16 MB  [4H][2H] bf16
    float* gates = (float*)(ws + (size_t)18 * (1 << 20));   //  8 MB  [B][4H] f32
    float* cst   = (float*)(ws + (size_t)26 * (1 << 20));   //  2 MB  [B][H] f32
    float* bias  = (float*)(ws + (size_t)28 * (1 << 20));   // 16 KB  [4H] f32

    hipLaunchKernelGGL(k_convert, dim3(32768), dim3(256), 0, stream,
                       W_ih, W_hh, b_ih, b_hh, Wcat, bias);
    hipLaunchKernelGGL(k_init, dim3((BATCH * HDIM) / 256), dim3(256), 0, stream,
                       noise, W_h0, b_h0, W_c0, b_c0, xh, cst);

    for (int s = 0; s < SEQ; ++s) {
        hipLaunchKernelGGL(k_gemm, dim3(N4H / 128, BATCH / 128), dim3(256), 0, stream,
                           xh, Wcat, gates);
        hipLaunchKernelGGL(k_cell, dim3(BATCH), dim3(256), 0, stream,
                           gates, bias, W_fc, b_fc, gumbel, dropm, s,
                           cst, xh, (float*)d_out);
    }
}
